// GraphAttentionLayer_39977555591548
// MI455X (gfx1250) — matrix operand order repaired and checked by reading, not compiled, not run on hardware
//
#include <hip/hip_runtime.h>

#define N_NODES 8192
#define F_DIM   256
#define NEGV    (-9000000000000000.0f)

typedef __attribute__((ext_vector_type(16))) __bf16          v16bf;
typedef __attribute__((ext_vector_type(8)))  float           v8f;
typedef __attribute__((ext_vector_type(16))) unsigned short  v16us;
typedef __attribute__((ext_vector_type(8)))  unsigned short  v8us;

__device__ __forceinline__ unsigned short f2bf(float f) {
    unsigned int u = __float_as_uint(f);
    u += 0x7fffu + ((u >> 16) & 1u);   // round-to-nearest-even
    return (unsigned short)(u >> 16);
}

__device__ __forceinline__ v16bf bf16v(v16us u) {
    return __builtin_bit_cast(v16bf, u);
}

// CDNA5 async DMA: global -> LDS, 16B per lane, GV addressing (ISA 15.18.3)
__device__ __forceinline__ void async_b128_to_lds(const void* gsrc, unsigned lds_off) {
    unsigned long long ga = (unsigned long long)gsrc;
    asm volatile("global_load_async_to_lds_b128 %0, %1, off"
                 :: "v"(lds_off), "v"(ga) : "memory");
}
__device__ __forceinline__ void wait_async0() {
    asm volatile("s_wait_asynccnt 0x0" ::: "memory");
}

// ---------------------------------------------------------------------------
// Kernel 1: W (f32, [K=256][N=256] row-major) -> W^T bf16 ([N][K])
// ---------------------------------------------------------------------------
__global__ void k_convW(const float* __restrict__ W, unsigned short* __restrict__ Wt) {
    int idx = blockIdx.x * 256 + threadIdx.x;          // 65536 elements
    int k = idx >> 8, n = idx & 255;
    Wt[n * F_DIM + k] = f2bf(W[k * F_DIM + n]);
}

// ---------------------------------------------------------------------------
// Kernel 2: Wh = h @ W via v_wmma_f32_16x16x32_bf16.
// One wave per 16x16 output tile; 8 waves per block.
// Writes Wh (f32 row-major) and Wh^T (bf16 [F][N]) for the attention GEMM.
// ---------------------------------------------------------------------------
__global__ void __launch_bounds__(256)
k_whgemm(const float* __restrict__ h, const unsigned short* __restrict__ Wt,
         float* __restrict__ Wh, unsigned short* __restrict__ WhT) {
    int lane = threadIdx.x & 31;
    int wid  = blockIdx.x * 8 + (threadIdx.x >> 5);    // 8192 tiles total
    int tM = wid >> 4, tN = wid & 15;
    int hlf = lane >> 4;                               // lane half (ISA A/B frag layout)
    int m = tM * 16 + (lane & 15);
    int n = tN * 16 + (lane & 15);

    const float*          hrow = h  + (size_t)m * F_DIM;
    const unsigned short* wcol = Wt + (size_t)n * F_DIM;

    v8f acc = {};
    for (int k = 0; k < F_DIM; k += 32) {
        // A fragment: row m, K = {hlf*8..+7} U {16+hlf*8..+7}   (ISA 7.12.2)
        float4 x0 = *(const float4*)(hrow + k + hlf * 8);
        float4 x1 = *(const float4*)(hrow + k + hlf * 8 + 4);
        float4 x2 = *(const float4*)(hrow + k + 16 + hlf * 8);
        float4 x3 = *(const float4*)(hrow + k + 16 + hlf * 8 + 4);
        v16us au;
        au[0]=f2bf(x0.x); au[1]=f2bf(x0.y); au[2]=f2bf(x0.z); au[3]=f2bf(x0.w);
        au[4]=f2bf(x1.x); au[5]=f2bf(x1.y); au[6]=f2bf(x1.z); au[7]=f2bf(x1.w);
        au[8]=f2bf(x2.x); au[9]=f2bf(x2.y); au[10]=f2bf(x2.z); au[11]=f2bf(x2.w);
        au[12]=f2bf(x3.x); au[13]=f2bf(x3.y); au[14]=f2bf(x3.z); au[15]=f2bf(x3.w);
        // B fragment: column n, K = {hlf*8..+7} U {16+hlf*8..+7} from W^T
        v8us blo = *(const v8us*)(wcol + k + hlf * 8);
        v8us bhi = *(const v8us*)(wcol + k + 16 + hlf * 8);
        v16us bu = __builtin_shufflevector(blo, bhi, 0,1,2,3,4,5,6,7,8,9,10,11,12,13,14,15);
        acc = __builtin_amdgcn_wmma_f32_16x16x32_bf16(
                  false, bf16v(au), false, bf16v(bu), (short)0, acc, false, false);
    }
#pragma unroll
    for (int r = 0; r < 8; ++r) {                      // C layout: M = r + hlf*8, N = lane%16
        int Mg = tM * 16 + r + hlf * 8;
        int Ng = tN * 16 + (lane & 15);
        float v = acc[r];
        Wh[(size_t)Mg * F_DIM + Ng]       = v;
        WhT[(size_t)Ng * N_NODES + Mg]    = f2bf(v);
    }
}

// ---------------------------------------------------------------------------
// Kernel 3: s1 = Wh@a1, s2 = Wh@a2.  One wave32 per row, shuffle reduce.
// ---------------------------------------------------------------------------
__global__ void k_s1s2(const float* __restrict__ Wh, const float* __restrict__ a,
                       float* __restrict__ s1, float* __restrict__ s2) {
    int lane = threadIdx.x & 31;
    int row  = blockIdx.x * 8 + (threadIdx.x >> 5);
    const float* wr = Wh + (size_t)row * F_DIM;
    float a1 = 0.f, a2 = 0.f;
#pragma unroll
    for (int t = 0; t < 8; ++t) {
        float w = wr[lane + 32 * t];
        a1 += w * a[lane + 32 * t];
        a2 += w * a[F_DIM + lane + 32 * t];
    }
#pragma unroll
    for (int off = 16; off > 0; off >>= 1) {           // wave32 reduction
        a1 += __shfl_xor(a1, off, 32);
        a2 += __shfl_xor(a2, off, 32);
    }
    if (lane == 0) { s1[row] = a1; s2[row] = a2; }
}

// ---------------------------------------------------------------------------
// Kernel 4: per-row online (max, sum-exp) over masked leaky-relu logits.
// One 256-thread block per row; streams 32 KB of adj per row as b128 loads.
// ---------------------------------------------------------------------------
__global__ void k_rowstats(const int* __restrict__ adj, const float* __restrict__ s1,
                           const float* __restrict__ s2,
                           float* __restrict__ mrow, float* __restrict__ linv) {
    __shared__ float rm[256], rl[256];
    int i = blockIdx.x, tid = threadIdx.x;
    float s1i = s1[i];
    const int4*   arow4 = (const int4*)(adj + (size_t)i * N_NODES);
    const float4* s24   = (const float4*)s2;
    float m = -3.4e38f, l = 0.f;
#pragma unroll
    for (int j4 = tid; j4 < N_NODES / 4; j4 += 256) {  // 8 iterations, b128 per lane
        int4   av = arow4[j4];
        float4 sv = s24[j4];
        float ev[4];
        ev[0] = s1i + sv.x; ev[1] = s1i + sv.y; ev[2] = s1i + sv.z; ev[3] = s1i + sv.w;
        int   am[4] = {av.x, av.y, av.z, av.w};
#pragma unroll
        for (int q = 0; q < 4; ++q) {
            float e = ev[q];
            e = e > 0.f ? e : 0.2f * e;                // leaky_relu(0.2)
            e = am[q] > 0 ? e : NEGV;
            if (e > m) { l *= __expf(m - e); m = e; }
            l += __expf(e - m);
        }
    }
    rm[tid] = m; rl[tid] = l;
    __syncthreads();
    for (int s = 128; s > 0; s >>= 1) {
        if (tid < s) {
            float m2 = rm[tid + s], l2 = rl[tid + s];
            float M = fmaxf(rm[tid], m2);
            rl[tid] = rl[tid] * __expf(rm[tid] - M) + l2 * __expf(m2 - M);
            rm[tid] = M;
        }
        __syncthreads();
    }
    if (tid == 0) { mrow[i] = rm[0]; linv[i] = 1.f / rl[0]; }
}

// ---------------------------------------------------------------------------
// Kernel 5: out = softmax(e) @ Wh, fused flash-style.
// Block = 8 waves, 16 output rows x 256 cols.  Per 128-col chunk:
//   adj tile (16x128 i32, 8KB) staged by async DMA into LDS (pipelined),
//   P tile (16x128 bf16) built in LDS, 8x v_wmma_f32_16x16x32_bf16 per wave.
// ---------------------------------------------------------------------------
__global__ void __launch_bounds__(256)
k_attn(const int* __restrict__ adj, const float* __restrict__ s2,
       const float* __restrict__ s1, const float* __restrict__ mrow,
       const float* __restrict__ linv, const unsigned short* __restrict__ WhT,
       float* __restrict__ out) {
    __shared__ alignas(16) int            s_adj[16 * 128];   // 8 KB async-DMA target
    __shared__ alignas(16) unsigned short s_p[16 * 136];     // 272B stride: conflict-free b128
    __shared__ float s_s1[16], s_m[16], s_li[16];

    int tid = threadIdx.x;
    int lane = tid & 31, wid = tid >> 5;
    int rowBase = blockIdx.x * 16;
    if (tid < 16) {
        s_s1[tid] = s1[rowBase + tid];
        s_m[tid]  = mrow[rowBase + tid];
        s_li[tid] = linv[rowBase + tid];
    }

    unsigned sadj_base = (unsigned)(uintptr_t)(void*)s_adj;  // low 32b of flat = LDS offset
    int o0 = tid * 16;                                       // this thread's 2 x 16B slots
    int o1 = (tid + 256) * 16;
    const int* g0 = adj + (size_t)(rowBase + (o0 >> 9)) * N_NODES + ((o0 & 511) >> 2);
    const int* g1 = adj + (size_t)(rowBase + (o1 >> 9)) * N_NODES + ((o1 & 511) >> 2);

    int hlf = lane >> 4;
    int n0 = wid * 32 + (lane & 15);
    int n1 = n0 + 16;
    const unsigned short* b0p = WhT + (size_t)n0 * N_NODES + hlf * 8;
    const unsigned short* b1p = WhT + (size_t)n1 * N_NODES + hlf * 8;
    int mA = lane & 15;
    int cc = tid & 127, rb = tid >> 7;                       // p-build coords

    // prologue: DMA first adj tile
    async_b128_to_lds(g0, sadj_base + o0);
    async_b128_to_lds(g1, sadj_base + o1);

    v8f acc0 = {}, acc1 = {};
    for (int jc = 0; jc < N_NODES; jc += 128) {
        wait_async0();
        __syncthreads();                                     // adj tile visible block-wide

        // Build P tile: 2048 elems / 256 threads (col fixed per thread)
        float s2v = s2[jc + cc];
#pragma unroll
        for (int q = 0; q < 8; ++q) {
            int r = rb + q * 2;
            float e = s_s1[r] + s2v;
            e = e > 0.f ? e : 0.2f * e;
            e = s_adj[r * 128 + cc] > 0 ? e : NEGV;
            s_p[r * 136 + cc] = f2bf(__expf(e - s_m[r]));
        }
        __syncthreads();                                     // P ready, s_adj free

        if (jc + 128 < N_NODES) {                            // DMA next tile under the WMMAs
            async_b128_to_lds(g0 + (jc + 128), sadj_base + o0);
            async_b128_to_lds(g1 + (jc + 128), sadj_base + o1);
        }

#pragma unroll
        for (int kb = 0; kb < 4; ++kb) {
            // A fragment from LDS: row mA, K = kb*32 + {hlf*8..+7} U {16+hlf*8..+7}
            v8us lo = *(const v8us*)&s_p[mA * 136 + kb * 32 + hlf * 8];
            v8us hi = *(const v8us*)&s_p[mA * 136 + kb * 32 + 16 + hlf * 8];
            v16us au = __builtin_shufflevector(lo, hi, 0,1,2,3,4,5,6,7,8,9,10,11,12,13,14,15);
            v16bf aa = bf16v(au);
            // B fragments from L2-resident Wh^T (K = {hlf*8..+7} U {16+hlf*8..+7})
            v16us b0 = __builtin_shufflevector(*(const v8us*)(b0p + jc + kb * 32), *(const v8us*)(b0p + jc + kb * 32 + 16), 0,1,2,3,4,5,6,7,8,9,10,11,12,13,14,15);
            v16us b1 = __builtin_shufflevector(*(const v8us*)(b1p + jc + kb * 32), *(const v8us*)(b1p + jc + kb * 32 + 16), 0,1,2,3,4,5,6,7,8,9,10,11,12,13,14,15);
            acc0 = __builtin_amdgcn_wmma_f32_16x16x32_bf16(
                       false, aa, false, bf16v(b0), (short)0, acc0, false, false);
            acc1 = __builtin_amdgcn_wmma_f32_16x16x32_bf16(
                       false, aa, false, bf16v(b1), (short)0, acc1, false, false);
        }
    }
#pragma unroll
    for (int r = 0; r < 8; ++r) {
        int M = r + hlf * 8;
        float sc = s_li[M];
        out[(size_t)(rowBase + M) * F_DIM + n0] = acc0[r] * sc;
        out[(size_t)(rowBase + M) * F_DIM + n1] = acc1[r] * sc;
    }
}

// ---------------------------------------------------------------------------
extern "C" void kernel_launch(void* const* d_in, const int* in_sizes, int n_in,
                              void* d_out, int out_size, void* d_ws, size_t ws_size,
                              hipStream_t stream) {
    const float* h   = (const float*)d_in[0];   // [8192][256]
    const int*   adj = (const int*)d_in[1];     // [8192][8192]
    const float* W   = (const float*)d_in[2];   // [256][256]
    const float* a   = (const float*)d_in[3];   // [512]

    char* ws = (char*)d_ws;
    size_t off = 0;
    float*          Wh  = (float*)(ws + off);          off += (size_t)N_NODES * F_DIM * sizeof(float);          // 8 MB
    unsigned short* WhT = (unsigned short*)(ws + off); off += (size_t)F_DIM * N_NODES * sizeof(unsigned short); // 4 MB
    unsigned short* Wt  = (unsigned short*)(ws + off); off += (size_t)F_DIM * F_DIM * sizeof(unsigned short);   // 128 KB
    float*          s1  = (float*)(ws + off);          off += N_NODES * sizeof(float);
    float*          s2  = (float*)(ws + off);          off += N_NODES * sizeof(float);
    float*          mr  = (float*)(ws + off);          off += N_NODES * sizeof(float);
    float*          li  = (float*)(ws + off);          off += N_NODES * sizeof(float);
    float* out = (float*)d_out;

    k_convW  <<<(F_DIM * F_DIM) / 256, 256, 0, stream>>>(W, Wt);
    k_whgemm <<<(N_NODES / 16) * (F_DIM / 16) / 8, 256, 0, stream>>>(h, Wt, Wh, WhT);
    k_s1s2   <<<N_NODES / 8, 256, 0, stream>>>(Wh, a, s1, s2);
    k_rowstats<<<N_NODES, 256, 0, stream>>>(adj, s1, s2, mr, li);
    k_attn   <<<N_NODES / 16, 256, 0, stream>>>(adj, s2, s1, mr, li, WhT, out);
}
